// RNNModel_61057255080231
// MI455X (gfx1250) — matrix operand order repaired and checked by reading, not compiled, not run on hardware
//
#include <hip/hip_runtime.h>
#include <hip/hip_bf16.h>
#include <math.h>

// ---------------- problem constants ----------------
#define TT   128      // time steps
#define BB   64       // batch
#define DD   1024     // embed dim
#define HH   1024     // hidden
#define HA_  128      // aux hidden
#define VV   10000    // vocab
#define MR   (TT*BB)  // 8192 rows after flattening time*batch

typedef __bf16 bhalf;
typedef __attribute__((ext_vector_type(8)))  float v8f;
typedef __attribute__((ext_vector_type(8)))  bhalf v8bh;
typedef __attribute__((ext_vector_type(16))) bhalf v16bh;

#define WMMA_BF16(a, b, c) \
  __builtin_amdgcn_wmma_f32_16x16x32_bf16(false, (a), false, (b), (short)0, (c), false, false)

// ---------------- helpers ----------------
__device__ __forceinline__ unsigned short f2bf(float f) {
  unsigned int u = __float_as_uint(f);
  u += 0x7FFFu + ((u >> 16) & 1u);       // round-to-nearest-even
  return (unsigned short)(u >> 16);
}
__device__ __forceinline__ void store_bf16(bhalf* p, float v) {
  *(unsigned short*)p = f2bf(v);
}
__device__ __forceinline__ float sigmoidf_(float z) {
  return 1.0f / (1.0f + expf(-z));
}

// A fragment base pointer for this lane (16x32 bf16 tile layout, row-major A).
// lanes 0-15: row=lane, K = k0+{0..15}
// lanes 16-31: row=lane-16, K = k0+{16..31}
__device__ __forceinline__ const bhalf* a_base(const bhalf* __restrict__ A, int lda,
                                               int row0, int lane) {
  int r   = lane & 15;
  int off = (lane < 16) ? 0 : 16;
  return A + (size_t)(row0 + r) * lda + off;
}
__device__ __forceinline__ v16bh load_a(const bhalf* __restrict__ p) {
  v8bh lo = *(const v8bh*)p;
  v8bh hi = *(const v8bh*)(p + 8);
  return __builtin_shufflevector(lo, hi, 0,1,2,3,4,5,6,7,8,9,10,11,12,13,14,15);
}

// B fragment base pointer (32x16 bf16 tile from N-major transposed weight Bt[N][K]).
// lanes 0-15: col=lane, K=k0..k0+15 ; lanes 16-31: col=lane-16, K=k0+16..k0+31
__device__ __forceinline__ const bhalf* b_base(const bhalf* __restrict__ Bt, int ldb,
                                               int col0, int lane) {
  int c   = lane & 15;
  int off = (lane < 16) ? 0 : 16;
  return Bt + (size_t)(col0 + c) * ldb + off;
}
__device__ __forceinline__ v16bh load_b(const bhalf* __restrict__ p) {
  return *(const v16bh*)p;
}

// ---------------- setup kernels ----------------

// dst[n*K + k] = bf16(src[k*N + n])   (transpose + convert)
__global__ void k_cvt_t(const float* __restrict__ src, bhalf* __restrict__ dst,
                        int K, int N) {
  size_t idx = (size_t)blockIdx.x * blockDim.x + threadIdx.x;
  if (idx >= (size_t)K * N) return;
  int n = (int)(idx / K);
  int k = (int)(idx % K);
  store_bf16(dst + idx, src[(size_t)k * N + n]);
}

__global__ void k_zero_u16(unsigned short* __restrict__ p, int n) {
  int i = blockIdx.x * blockDim.x + threadIdx.x;
  if (i < n) p[i] = 0;
}

// x_bf16[(t*B+b)][d] = bf16(emb[tokens[t*B+b]][d])
__global__ void k_embed(const int* __restrict__ tokens, const float* __restrict__ emb,
                        bhalf* __restrict__ x) {
  int rb = blockIdx.x;                    // 0 .. MR-1
  int tok = tokens[rb];
  const float* src = emb + (size_t)tok * DD;
  bhalf* dst = x + (size_t)rb * DD;
  for (int i = threadIdx.x; i < DD; i += blockDim.x) store_bf16(dst + i, src[i]);
}

__global__ void k_copy_f32(const float* __restrict__ s, float* __restrict__ d, int n) {
  int i = blockIdx.x * blockDim.x + threadIdx.x;
  if (i < n) d[i] = s[i];
}

// ---------------- generic bf16 WMMA GEMM: C = A(MxK) * Bt(NxK)^T + bias ----------------
// block: 256 threads = 8 waves; wave -> 32x64 tile (2 M-frags x 4 N-frags, 8 accums).
// grid: x = ceil(N/64), y = M/256.  Requires K%32==0, M%256==0, N%16==0, N>=64.
// Ragged N handled by overlap-clamping col0 (duplicate blocks recompute identical data).
__global__ __launch_bounds__(256)
void k_gemm_bf16(const bhalf* __restrict__ A, const bhalf* __restrict__ Bt,
                 const float* __restrict__ bias, float* __restrict__ C,
                 int M, int N, int K) {
  int lane = threadIdx.x & 31;
  int wave = threadIdx.x >> 5;
  int row0 = (blockIdx.y * 8 + wave) * 32;
  int col0 = blockIdx.x * 64;
  if (col0 + 64 > N) col0 = N - 64;     // wave-uniform overlap clamp
  if (row0 >= M) return;

  const bhalf* ap0 = a_base(A, K, row0,      lane);
  const bhalf* ap1 = a_base(A, K, row0 + 16, lane);
  const bhalf* bp0 = b_base(Bt, K, col0 +  0, lane);
  const bhalf* bp1 = b_base(Bt, K, col0 + 16, lane);
  const bhalf* bp2 = b_base(Bt, K, col0 + 32, lane);
  const bhalf* bp3 = b_base(Bt, K, col0 + 48, lane);

  v8f c00 = {}, c01 = {}, c02 = {}, c03 = {};
  v8f c10 = {}, c11 = {}, c12 = {}, c13 = {};
  for (int k0 = 0; k0 < K; k0 += 32) {
    v16bh a0 = load_a(ap0 + k0);
    v16bh a1 = load_a(ap1 + k0);
    __builtin_prefetch(ap0 + k0 + 32, 0, 3);   // speculative; dropped if OOB
    __builtin_prefetch(ap1 + k0 + 32, 0, 3);
    v16bh b0 = load_b(bp0 + k0);
    c00 = WMMA_BF16(a0, b0, c00);
    c10 = WMMA_BF16(a1, b0, c10);
    v16bh b1 = load_b(bp1 + k0);
    c01 = WMMA_BF16(a0, b1, c01);
    c11 = WMMA_BF16(a1, b1, c11);
    v16bh b2 = load_b(bp2 + k0);
    c02 = WMMA_BF16(a0, b2, c02);
    c12 = WMMA_BF16(a1, b2, c12);
    v16bh b3 = load_b(bp3 + k0);
    c03 = WMMA_BF16(a0, b3, c03);
    c13 = WMMA_BF16(a1, b3, c13);
  }

  int c  = lane & 15;
  float bi0 = bias[col0 +  0 + c];
  float bi1 = bias[col0 + 16 + c];
  float bi2 = bias[col0 + 32 + c];
  float bi3 = bias[col0 + 48 + c];
  int rbo = (lane < 16) ? 0 : 8;

  float* crow0 = C + (size_t)(row0 + rbo) * N + col0 + c;
#pragma unroll
  for (int i = 0; i < 8; i++) {
    crow0[ 0] = c00[i] + bi0;
    crow0[16] = c01[i] + bi1;
    crow0[32] = c02[i] + bi2;
    crow0[48] = c03[i] + bi3;
    crow0 += N;
  }
  float* crow1 = C + (size_t)(row0 + 16 + rbo) * N + col0 + c;
#pragma unroll
  for (int i = 0; i < 8; i++) {
    crow1[ 0] = c10[i] + bi0;
    crow1[16] = c11[i] + bi1;
    crow1[32] = c12[i] + bi2;
    crow1[48] = c13[i] + bi3;
    crow1 += N;
  }
}

// ---------------- per-step gate kernel (1 workgroup, 4 waves) ----------------
// h0n = tanh(ax_t + h0p @ Wah) ; gates = sigmoid(h0n @ Wa + ba)
__global__ __launch_bounds__(128)
void k_gate(const bhalf* __restrict__ h0p, const bhalf* __restrict__ WahT,
            const float* __restrict__ ax_t, const float* __restrict__ Wa,
            const float* __restrict__ ba, bhalf* __restrict__ h0n,
            float* __restrict__ gates) {
  __shared__ float sh[BB * HA_];
  int lane = threadIdx.x & 31;
  int wave = threadIdx.x >> 5;            // 0..3 -> M-tile
  int row0 = wave * 16;

  const bhalf* ap = a_base(h0p, HA_, row0, lane);

  for (int tn = 0; tn < 8; tn++) {        // 8 N-tiles of 16 over HA=128
    int col0 = tn * 16;
    const bhalf* bp = b_base(WahT, HA_, col0, lane);
    v8f acc = {};
#pragma unroll
    for (int k0 = 0; k0 < HA_; k0 += 32)
      acc = WMMA_BF16(load_a(ap + k0), load_b(bp + k0), acc);

    int c  = lane & 15;
    int rb = row0 + ((lane < 16) ? 0 : 8);
#pragma unroll
    for (int i = 0; i < 8; i++) {
      int m = rb + i;
      float v = tanhf(acc[i] + ax_t[(size_t)m * HA_ + col0 + c]);
      sh[m * HA_ + col0 + c] = v;
      store_bf16(h0n + (size_t)m * HA_ + col0 + c, v);
    }
  }
  __syncthreads();

  // 128 threads -> 64 rows x 2 gate outputs
  int row = threadIdx.x >> 1;
  int cl  = threadIdx.x & 1;
  float sum = ba[cl];
  for (int j = 0; j < HA_; j++) sum += sh[row * HA_ + j] * Wa[j * 2 + cl];
  gates[row * 2 + cl] = sigmoidf_(sum);
}

// ---------------- per-step h1 kernel (16 WGs x 4 waves; wave -> 16x64 tile) ----------------
// pre = preX_t + h1p @ Whh ; h1n = (1-s)*tanh(n*pre) + s*sigmoid(n*pre)
__global__ __launch_bounds__(128)
void k_step_h1(const bhalf* __restrict__ h1p, const bhalf* __restrict__ WhhT,
               const float* __restrict__ preX_t, const float* __restrict__ gates,
               bhalf* __restrict__ h1n_bf, bhalf* __restrict__ hs_t,
               float* __restrict__ h1f) {
  int lane = threadIdx.x & 31;
  int wave = threadIdx.x >> 5;
  int idx  = blockIdx.x * 4 + wave;       // 0..63
  int row0 = (idx & 3) * 16;              // 4 M-tiles over 64 rows
  int col0 = (idx >> 2) * 64;             // 16 col-groups of 64 over 1024 cols

  const bhalf* ap  = a_base(h1p, HH, row0, lane);
  const bhalf* bp0 = b_base(WhhT, HH, col0 +  0, lane);
  const bhalf* bp1 = b_base(WhhT, HH, col0 + 16, lane);
  const bhalf* bp2 = b_base(WhhT, HH, col0 + 32, lane);
  const bhalf* bp3 = b_base(WhhT, HH, col0 + 48, lane);

  v8f c0 = {}, c1 = {}, c2 = {}, c3 = {};
  for (int k0 = 0; k0 < HH; k0 += 32) {
    v16bh af = load_a(ap + k0);
    __builtin_prefetch(ap + k0 + 32, 0, 3);
    c0 = WMMA_BF16(af, load_b(bp0 + k0), c0);
    c1 = WMMA_BF16(af, load_b(bp1 + k0), c1);
    c2 = WMMA_BF16(af, load_b(bp2 + k0), c2);
    c3 = WMMA_BF16(af, load_b(bp3 + k0), c3);
  }

  int c  = lane & 15;
  int rb = row0 + ((lane < 16) ? 0 : 8);
#pragma unroll
  for (int j = 0; j < 4; j++) {
    v8f acc = (j == 0) ? c0 : (j == 1) ? c1 : (j == 2) ? c2 : c3;
    int col = col0 + j * 16 + c;
#pragma unroll
    for (int i = 0; i < 8; i++) {
      int m = rb + i;
      float pre = acc[i] + preX_t[(size_t)m * HH + col];
      float n = gates[m * 2 + 0];
      float s = gates[m * 2 + 1];
      float z = n * pre;
      float h = (1.0f - s) * tanhf(z) + s * sigmoidf_(z);
      store_bf16(h1n_bf + (size_t)m * HH + col, h);
      store_bf16(hs_t   + (size_t)m * HH + col, h);
      h1f[(size_t)m * HH + col] = h;
    }
  }
}

// ---------------- host orchestration ----------------
extern "C" void kernel_launch(void* const* d_in, const int* in_sizes, int n_in,
                              void* d_out, int out_size, void* d_ws, size_t ws_size,
                              hipStream_t stream) {
  (void)in_sizes; (void)n_in; (void)out_size; (void)ws_size;

  const int*   tokens = (const int*)  d_in[0];
  const float* emb    = (const float*)d_in[1];
  const float* Wax    = (const float*)d_in[2];
  const float* Wah    = (const float*)d_in[3];
  const float* bah    = (const float*)d_in[4];
  const float* Wa     = (const float*)d_in[5];
  const float* ba     = (const float*)d_in[6];
  const float* Wih    = (const float*)d_in[7];
  const float* Whh    = (const float*)d_in[8];
  const float* bh     = (const float*)d_in[9];
  const float* Wd     = (const float*)d_in[10];
  const float* bd     = (const float*)d_in[11];

  float* out = (float*)d_out;
  float* dec = out;                                  // (T*B, V)
  float* h1_final = out + (size_t)MR * VV;           // (B, H)

  // workspace carve-up (256B aligned)
  char*  base = (char*)d_ws;
  size_t off  = 0;
  auto take = [&](size_t bytes) -> char* {
    char* p = base + off;
    off += (bytes + 255) & ~(size_t)255;
    return p;
  };
  bhalf* x_bf   = (bhalf*)take((size_t)MR * DD * 2);       // embedded inputs
  bhalf* WihT   = (bhalf*)take((size_t)HH * DD * 2);       // (H, D)
  bhalf* WhhT   = (bhalf*)take((size_t)HH * HH * 2);       // (H, H)
  bhalf* WdT    = (bhalf*)take((size_t)VV * HH * 2);       // (V, H)
  bhalf* WaxT   = (bhalf*)take((size_t)HA_ * DD * 2);      // (HA, D)
  bhalf* WahT   = (bhalf*)take((size_t)HA_ * HA_ * 2);     // (HA, HA)
  float* preX   = (float*)take((size_t)MR * HH * 4);       // x@Wih + bh
  float* ax     = (float*)take((size_t)MR * HA_ * 4);      // x@Wax + bah
  bhalf* hs_bf  = (bhalf*)take((size_t)MR * HH * 2);       // all h1n, bf16
  bhalf* h0bf[2] = { (bhalf*)take((size_t)BB * HA_ * 2),
                     (bhalf*)take((size_t)BB * HA_ * 2) };
  bhalf* h1bf[2] = { (bhalf*)take((size_t)BB * HH * 2),
                     (bhalf*)take((size_t)BB * HH * 2) };
  float* h1f    = (float*)take((size_t)BB * HH * 4);
  float* gates  = (float*)take((size_t)BB * 2 * 4);

  // 1) convert + transpose weights to bf16 N-major
  {
    size_t n;
    n = (size_t)DD * HH;
    k_cvt_t<<<dim3((unsigned)((n + 255) / 256)), 256, 0, stream>>>(Wih, WihT, DD, HH);
    n = (size_t)HH * HH;
    k_cvt_t<<<dim3((unsigned)((n + 255) / 256)), 256, 0, stream>>>(Whh, WhhT, HH, HH);
    n = (size_t)HH * VV;
    k_cvt_t<<<dim3((unsigned)((n + 255) / 256)), 256, 0, stream>>>(Wd, WdT, HH, VV);
    n = (size_t)DD * HA_;
    k_cvt_t<<<dim3((unsigned)((n + 255) / 256)), 256, 0, stream>>>(Wax, WaxT, DD, HA_);
    n = (size_t)HA_ * HA_;
    k_cvt_t<<<dim3((unsigned)((n + 255) / 256)), 256, 0, stream>>>(Wah, WahT, HA_, HA_);
  }

  // 2) zero recurrent state
  k_zero_u16<<<dim3((BB * HA_ + 255) / 256), 256, 0, stream>>>((unsigned short*)h0bf[0], BB * HA_);
  k_zero_u16<<<dim3((BB * HA_ + 255) / 256), 256, 0, stream>>>((unsigned short*)h0bf[1], BB * HA_);
  k_zero_u16<<<dim3((BB * HH  + 255) / 256), 256, 0, stream>>>((unsigned short*)h1bf[0], BB * HH);
  k_zero_u16<<<dim3((BB * HH  + 255) / 256), 256, 0, stream>>>((unsigned short*)h1bf[1], BB * HH);

  // 3) embedding gather -> bf16
  k_embed<<<dim3(MR), 256, 0, stream>>>(tokens, emb, x_bf);

  // 4) time-parallel precompute: preX = x@Wih + bh ; ax = x@Wax + bah
  k_gemm_bf16<<<dim3(HH / 64, MR / 256), 256, 0, stream>>>(x_bf, WihT, bh, preX, MR, HH, DD);
  k_gemm_bf16<<<dim3(HA_ / 64, MR / 256), 256, 0, stream>>>(x_bf, WaxT, bah, ax, MR, HA_, DD);

  // 5) sequential scan: 2 small kernels per step, ping-pong state
  for (int t = 0; t < TT; t++) {
    int pi = t & 1, ni = (t + 1) & 1;
    k_gate<<<dim3(1), 128, 0, stream>>>(h0bf[pi], WahT, ax + (size_t)t * BB * HA_,
                                        Wa, ba, h0bf[ni], gates);
    k_step_h1<<<dim3(16), 128, 0, stream>>>(h1bf[pi], WhhT, preX + (size_t)t * BB * HH,
                                            gates, h1bf[ni], hs_bf + (size_t)t * BB * HH,
                                            h1f);
  }

  // 6) decoder: decoded = hs @ Wd + bd   (M=8192, N=10000, K=1024)
  k_gemm_bf16<<<dim3((VV + 63) / 64, MR / 256), 256, 0, stream>>>(hs_bf, WdT, bd, dec, MR, VV, HH);

  // 7) h1_final
  k_copy_f32<<<dim3((BB * HH + 255) / 256), 256, 0, stream>>>(h1f, h1_final, BB * HH);
}
